// VARnetReg_257698038002
// MI455X (gfx1250) — matrix operand order repaired and checked by reading, not compiled, not run on hardware
//
#include <hip/hip_runtime.h>
#include <math.h>

typedef __attribute__((ext_vector_type(16))) _Float16 v16h;
typedef __attribute__((ext_vector_type(8)))  _Float16 v8h;
typedef __attribute__((ext_vector_type(8)))  float    v8f;

#define B_    64
#define SEQ_  4096
#define NF_   32
#define LDWT  2050
#define KFFT  1026
#define KPAD  1056   /* rfft bins padded to multiple of 32 */
#define TPAD  2080   /* time length padded to multiple of 32 */
#define MDFT  (2*KPAD) /* 2112 rows: [cos | -sin] blocks */
#define PROJ_ 256

// ---------------------------------------------------------------- helpers
__device__ __forceinline__ float gelu_exact(float x) {
  return 0.5f * x * (1.0f + erff(x * 0.70710678118654752f));
}

__device__ __forceinline__ v16h join16(v8h a0, v8h a1) {
  return __builtin_shufflevector(a0, a1, 0,1,2,3,4,5,6,7,8,9,10,11,12,13,14,15);
}

// ---------------------------------------------------------------- repack kernels
__global__ void convert_x_kernel(const float* __restrict__ x, _Float16* __restrict__ o, long n) {
  long i = (long)blockIdx.x * 256 + threadIdx.x;
  if (i < n) o[i] = (_Float16)x[i];
}

// w: [OC][IC][K] f32 -> wp: [K][OCp][ICp] f16, zero padded
__global__ void repack_w_kernel(const float* __restrict__ w, _Float16* __restrict__ wp,
                                int OC, int IC, int K, int OCp, int ICp) {
  long idx = (long)blockIdx.x * 256 + threadIdx.x;
  long n = (long)K * OCp * ICp;
  if (idx >= n) return;
  int ic = (int)(idx % ICp);
  int oc = (int)((idx / ICp) % OCp);
  int k  = (int)(idx / ((long)ICp * OCp));
  float v = 0.f;
  if (oc < OC && ic < IC) v = w[((long)oc * IC + ic) * K + k];
  wp[idx] = (_Float16)v;
}

// U [256][1026] f32 -> Uf [256][KPAD] f16, zero pad
__global__ void convert_U_kernel(const float* __restrict__ U, _Float16* __restrict__ Uf) {
  long idx = (long)blockIdx.x * 256 + threadIdx.x;
  if (idx >= (long)PROJ_ * KPAD) return;
  int k = (int)(idx % KPAD);
  int m = (int)(idx / KPAD);
  Uf[idx] = (_Float16)((k < KFFT) ? U[(long)m * KFFT + k] : 0.f);
}

// Trig [MDFT][TPAD]: row m = ri*KPAD + k; val = (ri?-sin:cos)(2*pi*k*t/2050)/sqrt(2050)
__global__ void make_trig_kernel(_Float16* __restrict__ trig) {
  long idx = (long)blockIdx.x * 256 + threadIdx.x;
  if (idx >= (long)MDFT * TPAD) return;
  int t = (int)(idx % TPAD);
  int m = (int)(idx / TPAD);
  int ri = m / KPAD;
  int k  = m % KPAD;
  float val = 0.f;
  if (k < KFFT && t < LDWT) {
    int r = (int)(((long long)k * (long long)t) % LDWT);  // exact angle reduction
    float ang = 6.283185307179586f * ((float)r / (float)LDWT);
    float sv = sinf(ang), cv = cosf(ang);
    val = ((ri == 0) ? cv : -sv) * 0.02208630521f;        // 1/sqrt(2050)
  }
  trig[idx] = (_Float16)val;
}

// act6 [B][2050][32] -> h6T [B*32 cols][TPAD] (K=t contiguous), zero pad t>=2050
__global__ void repack_h6_kernel(const _Float16* __restrict__ a6, _Float16* __restrict__ h6t) {
  long idx = (long)blockIdx.x * 256 + threadIdx.x;
  if (idx >= (long)B_ * 32 * TPAD) return;
  int t = (int)(idx % TPAD);
  int c = (int)((idx / TPAD) % 32);
  int b = (int)(idx / ((long)TPAD * 32));
  _Float16 v = (_Float16)0.f;
  if (t < LDWT) v = a6[((long)b * LDWT + t) * 32 + c];
  h6t[idx] = v;
}

// ---------------------------------------------------------------- conv1d: LDS-staged shifted WMMA GEMM + bias + GELU
// in: [B][L][ICp] f16, wp: [K][OCp][ICp] f16, out: [B][L][OCp] f16
// Block: 8 waves stage positions [tn0*16-pad, tn0*16+127+pad] x ICp into LDS once,
// then each wave computes its 16-position column strip for ALL OC tiles.
__global__ __launch_bounds__(256) void conv1d_wmma_gelu(
    const _Float16* __restrict__ in, const _Float16* __restrict__ wp,
    const float* __restrict__ bias, _Float16* __restrict__ out,
    int L, int ICp, int OC, int OCp, int K, int pad, int tilesN) {
  extern __shared__ _Float16 smem[];          // [span][ICp+8] (row-padded vs bank conflicts)
  const int tn0 = blockIdx.x * 8;
  const int b   = blockIdx.z;
  const int p0  = tn0 * 16 - pad;
  const int span = 128 + 2 * pad;
  const int slds = ICp + 8;

  // cooperative stage (zero-filled outside [0,L))
  {
    int chunks = ICp >> 3;                    // 8-half vectors per row
    int totalVec = span * chunks;
    for (int vi = threadIdx.x; vi < totalVec; vi += 256) {
      int prel = vi / chunks;
      int cc = (vi - prel * chunks) << 3;
      int pos = p0 + prel;
      v8h val = {};
      if (pos >= 0 && pos < L) val = *(const v8h*)(in + ((long)b * L + pos) * ICp + cc);
      *(v8h*)(smem + (long)prel * slds + cc) = val;
    }
  }
  __syncthreads();

  const int lane = threadIdx.x & 31;
  const int wv   = threadIdx.x >> 5;
  const int tn   = tn0 + wv;                  // wave-uniform
  if (tn >= tilesN) return;
  const int g = lane >> 4, l16 = lane & 15;
  const int posN = tn * 16 + l16;
  const int srow0 = wv * 16 + l16;            // smem row for tap k is srow0 + k

  for (int tm = 0; tm < OCp / 16; ++tm) {
    v8f acc = {};
    for (int k = 0; k < K; ++k) {
      const _Float16* brow = smem + (long)(srow0 + k) * slds;
      const _Float16* arow = wp + ((long)k * OCp + tm * 16 + l16) * ICp;
      for (int ic0 = 0; ic0 < ICp; ic0 += 32) {
        v16h A = join16(*(const v8h*)(arow + ic0 + g * 8),
                        *(const v8h*)(arow + ic0 + 16 + g * 8));
        v16h Bv = join16(*(const v8h*)(brow + ic0 + g * 8),
                         *(const v8h*)(brow + ic0 + 16 + g * 8));
        acc = __builtin_amdgcn_wmma_f32_16x16x32_f16(false, A, false, Bv,
                                                     (short)0, acc, false, false);
      }
    }
    if (posN < L) {
      int mbase = tm * 16 + g * 8;            // D rows: lanes<16 -> M=v, lanes>=16 -> M=v+8
      v8h o;
      #pragma unroll
      for (int v = 0; v < 8; ++v) {
        int m = mbase + v;
        float xv = acc[v] + ((m < OC) ? bias[m] : 0.f);
        o[v] = (_Float16)gelu_exact(xv);
      }
      *(v8h*)(out + ((long)b * L + posN) * OCp + mbase) = o;
    }
  }
}

// ---------------------------------------------------------------- DWT(bior2.2,symmetric) + linear resize -> cat[B][2050][192]
__global__ void dwt_interp_kernel(const _Float16* __restrict__ h3, _Float16* __restrict__ cat) {
  long idx = (long)blockIdx.x * 256 + threadIdx.x;
  if (idx >= (long)B_ * LDWT * 64) return;
  int c = (int)(idx & 63);
  int i = (int)((idx >> 6) % LDWT);
  int b = (int)((idx >> 6) / LDWT);
  const _Float16* xp = h3 + (long)b * SEQ_ * 64 + c;

  // linear interpolate 4096 -> 2050 (align_corners=False)
  float src = (i + 0.5f) * ((float)SEQ_ / (float)LDWT) - 0.5f;
  float fl = floorf(src);
  float w = src - fl;
  int i0 = (int)fl;
  int i0c = i0 < 0 ? 0 : (i0 > SEQ_ - 1 ? SEQ_ - 1 : i0);
  int i1c = (i0 + 1) < 0 ? 0 : ((i0 + 1) > SEQ_ - 1 ? SEQ_ - 1 : (i0 + 1));
  float ds = (1.f - w) * (float)xp[(long)i0c * 64] + w * (float)xp[(long)i1c * 64];

  // dwt: y[i] = sum_j f[j] * x_sym(2i+1-j)
  const float LO[6] = {0.f, -0.17677669529663689f, 0.35355339059327379f,
                       1.0606601717798214f, 0.35355339059327379f, -0.17677669529663689f};
  const float HI[6] = {0.f, 0.35355339059327379f, -0.70710678118654757f,
                       0.35355339059327379f, 0.f, 0.f};
  float a = 0.f, d = 0.f;
  #pragma unroll
  for (int j = 0; j < 6; ++j) {
    int q = 2 * i + 1 - j;
    int t = (q < 0) ? (-q - 1) : ((q >= SEQ_) ? (2 * SEQ_ - 1 - q) : q);
    float xv = (float)xp[(long)t * 64];
    a += LO[j] * xv;
    d += HI[j] * xv;
  }
  _Float16* crow = cat + ((long)b * LDWT + i) * 192;
  crow[c]       = (_Float16)ds;
  crow[64 + c]  = (_Float16)a;
  crow[128 + c] = (_Float16)d;
}

// ---------------------------------------------------------------- 2x2-blocked GEMM cores
// A row-major [M][Ktot] (lda=Ktot), B K-contiguous per column [N][Ktot] (ldb=Ktot).
__device__ __forceinline__ void gemm_2x2(const _Float16* __restrict__ A0,
                                         const _Float16* __restrict__ A1,
                                         const _Float16* __restrict__ B0,
                                         const _Float16* __restrict__ B1,
                                         int Ktot, int g,
                                         v8f& c00, v8f& c01, v8f& c10, v8f& c11) {
  for (int k0 = 0; k0 < Ktot; k0 += 32) {
    if (k0 + 128 < Ktot) {
      __builtin_prefetch(A0 + k0 + 128, 0, 0);
      __builtin_prefetch(B0 + k0 + 128, 0, 0);
    }
    v16h a0 = *(const v16h*)(A0 + k0 + g * 16);
    v16h a1 = *(const v16h*)(A1 + k0 + g * 16);
    v16h b0 = *(const v16h*)(B0 + k0 + g * 16);
    v16h b1 = *(const v16h*)(B1 + k0 + g * 16);
    c00 = __builtin_amdgcn_wmma_f32_16x16x32_f16(false, a0, false, b0, (short)0, c00, false, false);
    c01 = __builtin_amdgcn_wmma_f32_16x16x32_f16(false, a0, false, b1, (short)0, c01, false, false);
    c10 = __builtin_amdgcn_wmma_f32_16x16x32_f16(false, a1, false, b0, (short)0, c10, false, false);
    c11 = __builtin_amdgcn_wmma_f32_16x16x32_f16(false, a1, false, b1, (short)0, c11, false, false);
  }
}

// ---------------------------------------------------------------- DFT GEMM: fr = Trig x h6T  (2x2 tiles per wave)
// trig [MDFT][TPAD], h6t [B*32][TPAD] -> fr [B][64][KPAD] (channel 2c+ri, k contiguous)
__device__ __forceinline__ void store_fr_tile(v8f acc, int tm, int tn, int g, int l16,
                                              _Float16* __restrict__ fr) {
  int n = tn * 16 + l16, b = n >> 5, c = n & 31;
  int mbase = tm * 16 + g * 8;
  int ri = mbase / KPAD, kk = mbase % KPAD;   // 8-row group never straddles ri blocks
  v8h o;
  #pragma unroll
  for (int v = 0; v < 8; ++v) o[v] = (_Float16)acc[v];
  *(v8h*)(fr + ((long)b * 64 + 2 * c + ri) * KPAD + kk) = o;
}

__global__ __launch_bounds__(256) void dft_gemm_kernel(const _Float16* __restrict__ trig,
                                                       const _Float16* __restrict__ h6t,
                                                       _Float16* __restrict__ fr) {
  int lane = threadIdx.x & 31, wv = threadIdx.x >> 5;
  int tnp = blockIdx.x * 8 + wv;   // 64 column-tile pairs (B*32 = 2048 cols)
  int tmp = blockIdx.y;            // 66 row-tile pairs (MDFT = 2112 rows)
  int g = lane >> 4, l16 = lane & 15;
  const _Float16* A0 = trig + ((long)tmp * 32 + l16) * TPAD;
  const _Float16* A1 = A0 + (long)16 * TPAD;
  const _Float16* B0 = h6t + ((long)tnp * 32 + l16) * TPAD;
  const _Float16* B1 = B0 + (long)16 * TPAD;
  v8f c00 = {}, c01 = {}, c10 = {}, c11 = {};
  gemm_2x2(A0, A1, B0, B1, TPAD, g, c00, c01, c10, c11);
  store_fr_tile(c00, 2 * tmp,     2 * tnp,     g, l16, fr);
  store_fr_tile(c01, 2 * tmp,     2 * tnp + 1, g, l16, fr);
  store_fr_tile(c10, 2 * tmp + 1, 2 * tnp,     g, l16, fr);
  store_fr_tile(c11, 2 * tmp + 1, 2 * tnp + 1, g, l16, fr);
}

// ---------------------------------------------------------------- Projection GEMM (2x2): act7[b][m][l] = sum_k Uf[m][k] * fr[b][l][k]
__device__ __forceinline__ void store_act7_tile(v8f acc, int tm, int tn, int g, int l16,
                                                _Float16* __restrict__ act7) {
  int n = tn * 16 + l16, b = n >> 6, l = n & 63;
  int mbase = tm * 16 + g * 8;
  #pragma unroll
  for (int v = 0; v < 8; ++v)
    act7[((long)b * PROJ_ + mbase + v) * 64 + l] = (_Float16)acc[v];
}

__global__ __launch_bounds__(256) void proj_gemm_kernel(const _Float16* __restrict__ Uf,
                                                        const _Float16* __restrict__ fr,
                                                        _Float16* __restrict__ act7) {
  int lane = threadIdx.x & 31, wv = threadIdx.x >> 5;
  int tnp = blockIdx.x * 8 + wv;   // 128 column-tile pairs (B*64 = 4096 cols)
  int tmp = blockIdx.y;            // 8 row-tile pairs (PROJ = 256 rows)
  int g = lane >> 4, l16 = lane & 15;
  const _Float16* A0 = Uf + ((long)tmp * 32 + l16) * KPAD;
  const _Float16* A1 = A0 + (long)16 * KPAD;
  const _Float16* B0 = fr + ((long)tnp * 32 + l16) * KPAD;
  const _Float16* B1 = B0 + (long)16 * KPAD;
  v8f c00 = {}, c01 = {}, c10 = {}, c11 = {};
  gemm_2x2(A0, A1, B0, B1, KPAD, g, c00, c01, c10, c11);
  store_act7_tile(c00, 2 * tmp,     2 * tnp,     g, l16, act7);
  store_act7_tile(c01, 2 * tmp,     2 * tnp + 1, g, l16, act7);
  store_act7_tile(c10, 2 * tmp + 1, 2 * tnp,     g, l16, act7);
  store_act7_tile(c11, 2 * tmp + 1, 2 * tnp + 1, g, l16, act7);
}

// ---------------------------------------------------------------- mean pool + head
__global__ void final_reduce_kernel(const _Float16* __restrict__ act8,
                                    const float* __restrict__ wh,
                                    const float* __restrict__ bh,
                                    float* __restrict__ out) {
  int b = blockIdx.x;
  int lane = threadIdx.x;  // 32 threads = channels
  float s = 0.f;
  for (int p = 0; p < PROJ_; ++p)
    s += (float)act8[((long)b * PROJ_ + p) * 32 + lane];
  float v = (s * (1.0f / (float)PROJ_)) * wh[lane];
  #pragma unroll
  for (int off = 16; off > 0; off >>= 1) v += __shfl_xor(v, off, 32);
  if (lane == 0) out[b] = v + bh[0];
}

// ---------------------------------------------------------------- launcher
extern "C" void kernel_launch(void* const* d_in, const int* in_sizes, int n_in,
                              void* d_out, int out_size, void* d_ws, size_t ws_size,
                              hipStream_t stream) {
  (void)in_sizes; (void)n_in; (void)out_size; (void)ws_size;
  const float* x  = (const float*)d_in[0];
  const float* w1 = (const float*)d_in[1];  const float* b1 = (const float*)d_in[2];
  const float* w2 = (const float*)d_in[3];  const float* b2 = (const float*)d_in[4];
  const float* w3 = (const float*)d_in[5];  const float* b3 = (const float*)d_in[6];
  const float* w4 = (const float*)d_in[7];  const float* b4 = (const float*)d_in[8];
  const float* w5 = (const float*)d_in[9];  const float* b5 = (const float*)d_in[10];
  const float* w6 = (const float*)d_in[11]; const float* b6 = (const float*)d_in[12];
  const float* U  = (const float*)d_in[13];
  const float* w7 = (const float*)d_in[14]; const float* b7 = (const float*)d_in[15];
  const float* w8 = (const float*)d_in[16]; const float* b8 = (const float*)d_in[17];
  const float* wh = (const float*)d_in[18]; const float* bh = (const float*)d_in[19];
  float* out = (float*)d_out;

  char* ws = (char*)d_ws;
  size_t off = 0;
  auto carve = [&](size_t bytes) -> void* {
    void* p = ws + off;
    off += (bytes + 255) & ~(size_t)255;
    return p;
  };
  // ping-pong activation regions (largest tenants: cat 50.4MB / h3 33.6MB)
  _Float16* regA = (_Float16*)carve((size_t)B_ * LDWT * 192 * 2);
  _Float16* regB = (_Float16*)carve((size_t)B_ * SEQ_ * 64 * 2);
  _Float16* trig = (_Float16*)carve((size_t)MDFT * TPAD * 2);
  _Float16* Uf   = (_Float16*)carve((size_t)PROJ_ * KPAD * 2);
  _Float16* wp1  = (_Float16*)carve((size_t)9 * 64 * 32 * 2);
  _Float16* wp2  = (_Float16*)carve((size_t)3 * 64 * 64 * 2);
  _Float16* wp3  = (_Float16*)carve((size_t)3 * 64 * 64 * 2);
  _Float16* wp4  = (_Float16*)carve((size_t)3 * 96 * 192 * 2);
  _Float16* wp5  = (_Float16*)carve((size_t)3 * 64 * 96 * 2);
  _Float16* wp6  = (_Float16*)carve((size_t)3 * 32 * 64 * 2);
  _Float16* wp7  = (_Float16*)carve((size_t)3 * 128 * 64 * 2);
  _Float16* wp8  = (_Float16*)carve((size_t)3 * 32 * 128 * 2);

  auto cdiv = [](long a, long b) -> unsigned { return (unsigned)((a + b - 1) / b); };
  auto shm  = [](int pad, int ICp) -> unsigned { return (unsigned)((128 + 2 * pad) * (ICp + 8) * 2); };

  // constant prep (independent)
  repack_w_kernel<<<cdiv(9L*64*32, 256), 256, 0, stream>>>(w1, wp1, 64, 32, 9, 64, 32);
  repack_w_kernel<<<cdiv(3L*64*64, 256), 256, 0, stream>>>(w2, wp2, 64, 64, 3, 64, 64);
  repack_w_kernel<<<cdiv(3L*64*64, 256), 256, 0, stream>>>(w3, wp3, 64, 64, 3, 64, 64);
  repack_w_kernel<<<cdiv(3L*96*192, 256), 256, 0, stream>>>(w4, wp4, 96, 192, 3, 96, 192);
  repack_w_kernel<<<cdiv(3L*64*96, 256), 256, 0, stream>>>(w5, wp5, 48, 96, 3, 64, 96);
  repack_w_kernel<<<cdiv(3L*32*64, 256), 256, 0, stream>>>(w6, wp6, 32, 48, 3, 32, 64);
  repack_w_kernel<<<cdiv(3L*128*64, 256), 256, 0, stream>>>(w7, wp7, 128, 64, 3, 128, 64);
  repack_w_kernel<<<cdiv(3L*32*128, 256), 256, 0, stream>>>(w8, wp8, 32, 128, 3, 32, 128);
  convert_U_kernel<<<cdiv((long)PROJ_*KPAD, 256), 256, 0, stream>>>(U, Uf);
  make_trig_kernel<<<cdiv((long)MDFT*TPAD, 256), 256, 0, stream>>>(trig);

  // x [B][4096][32] f32 -> act0 (same layout) f16 in regA
  convert_x_kernel<<<cdiv((long)B_*SEQ_*NF_, 256), 256, 0, stream>>>(x, regA, (long)B_*SEQ_*NF_);

  // conv1: 32 -> 64, k=9, L=4096    regA -> regB
  conv1d_wmma_gelu<<<dim3(32, 1, B_), 256, shm(4, 32), stream>>>(regA, wp1, b1, regB, SEQ_, 32, 64, 64, 9, 4, 256);
  // conv2: 64 -> 64, k=3            regB -> regA
  conv1d_wmma_gelu<<<dim3(32, 1, B_), 256, shm(1, 64), stream>>>(regB, wp2, b2, regA, SEQ_, 64, 64, 64, 3, 1, 256);
  // conv3: 64 -> 64, k=3            regA -> regB (= h3)
  conv1d_wmma_gelu<<<dim3(32, 1, B_), 256, shm(1, 64), stream>>>(regA, wp3, b3, regB, SEQ_, 64, 64, 64, 3, 1, 256);

  // DWT + resize concat: h3(regB) -> cat(regA) [B][2050][192]
  dwt_interp_kernel<<<cdiv((long)B_*LDWT*64, 256), 256, 0, stream>>>(regB, regA);

  // conv4: 192 -> 96, L=2050       regA -> regB   (129 position tiles)
  conv1d_wmma_gelu<<<dim3(17, 1, B_), 256, shm(1, 192), stream>>>(regA, wp4, b4, regB, LDWT, 192, 96, 96, 3, 1, 129);
  // conv5: 96 -> 48 (OCp=64)       regB -> regA
  conv1d_wmma_gelu<<<dim3(17, 1, B_), 256, shm(1, 96), stream>>>(regB, wp5, b5, regA, LDWT, 96, 48, 64, 3, 1, 129);
  // conv6: 48(pad64) -> 32         regA -> regB
  conv1d_wmma_gelu<<<dim3(17, 1, B_), 256, shm(1, 64), stream>>>(regA, wp6, b6, regB, LDWT, 64, 32, 32, 3, 1, 129);

  // channel-major repack for DFT: act6(regB) -> h6T(regA)
  repack_h6_kernel<<<cdiv((long)B_*32*TPAD, 256), 256, 0, stream>>>(regB, regA);

  // DFT GEMM (2x2): trig x h6T(regA) -> fr(regB) [B][64][KPAD]
  dft_gemm_kernel<<<dim3(8, MDFT / 32, 1), 256, 0, stream>>>(trig, regA, regB);

  // projection GEMM (2x2): Uf x fr(regB) -> act7(regA) [B][256][64]
  proj_gemm_kernel<<<dim3(16, PROJ_ / 32, 1), 256, 0, stream>>>(Uf, regB, regA);

  // conv7: 64 -> 128, L=256        regA -> regB
  conv1d_wmma_gelu<<<dim3(2, 1, B_), 256, shm(1, 64), stream>>>(regA, wp7, b7, regB, PROJ_, 64, 128, 128, 3, 1, 16);
  // conv8: 128 -> 32               regB -> regA
  conv1d_wmma_gelu<<<dim3(2, 1, B_), 256, shm(1, 128), stream>>>(regB, wp8, b8, regA, PROJ_, 128, 32, 32, 3, 1, 16);

  // mean pool + head
  final_reduce_kernel<<<dim3(B_), 32, 0, stream>>>(regA, wh, bh, out);
}
